// Qwen3MoeExperts_53042846105770
// MI455X (gfx1250) — matrix operand order repaired and checked by reading, not compiled, not run on hardware
//
#include <hip/hip_runtime.h>

// ---------------------------------------------------------------------------
// Qwen3 MoE expert MLP, fused, bf16x3 split-precision WMMA for gfx1250.
//   T=32768, H=2048, I=768.  out = (silu(X@Wg) * (X@Wu)) @ Wd
// ---------------------------------------------------------------------------

#define T_TOK  32768
#define HID    2048
#define INTER  768

typedef __bf16 bf16_t;
typedef bf16_t v16bf __attribute__((ext_vector_type(16)));
typedef float  v8f   __attribute__((ext_vector_type(8)));

union FragU {
    v16bf v;
    int4  q[2];
    unsigned u[8];
};

__device__ __forceinline__ unsigned short f2bf(float f) {
    unsigned u = __float_as_uint(f);
    u += 0x7FFFu + ((u >> 16) & 1u);          // round-to-nearest-even
    return (unsigned short)(u >> 16);
}
__device__ __forceinline__ float bf2f(unsigned short h) {
    return __uint_as_float(((unsigned)h) << 16);
}
__device__ __forceinline__ void split2(float f, unsigned short& hi, unsigned short& lo) {
    hi = f2bf(f);
    lo = f2bf(f - bf2f(hi));                  // residual captures next 8 mantissa bits
}

// ---------------------------------------------------------------------------
// Prologue: convert a row-major fp32 matrix [K x N] into pre-swizzled WMMA
// B-fragments (bf16 hi plane dwords 0..7, lo plane dwords 8..15 per lane).
// B layout (wave32, 16-bit): lanes 0-15 col n, dword j: K=2j,2j+1 (j<4), K=16+2(j-4),+1 (j>=4);
//                            lanes 16-31 same cols, same K +8.
// Record index: rec = (nt*kChunks + ks)*32 + lane ; record = 16 dwords.
// ---------------------------------------------------------------------------
__global__ void convert_b_frags(const float* __restrict__ src,
                                unsigned* __restrict__ dst,
                                int ldN, int kChunks, int totalRecs) {
    int rec = blockIdx.x * blockDim.x + threadIdx.x;
    if (rec >= totalRecs) return;
    int lane = rec & 31;
    int ks   = (rec >> 5) % kChunks;
    int nt   = rec / (32 * kChunks);
    int n     = nt * 16 + (lane & 15);
    int kbase = ks * 32 + ((lane >> 4) << 3);
    unsigned* out = dst + (size_t)rec * 16;
    #pragma unroll
    for (int j = 0; j < 8; ++j) {
        int k = kbase + ((j < 4) ? (2 * j) : (16 + 2 * (j - 4)));
        float a0 = src[(size_t)k * ldN + n];
        float a1 = src[(size_t)(k + 1) * ldN + n];
        unsigned short h0, l0, h1, l1;
        split2(a0, h0, l0);
        split2(a1, h1, l1);
        out[j]     = (unsigned)h0 | ((unsigned)h1 << 16);  // hi plane
        out[j + 8] = (unsigned)l0 | ((unsigned)l1 << 16);  // lo plane
    }
}

// ---------------------------------------------------------------------------
// Fused kernel: 32 tokens per block, 8 waves (256 threads).
// GEMM1 (X@Wgu): each wave owns 96 gate + 96 up columns (6 n-tiles of 16),
//   K staged cooperatively in LDS as A-fragments, bf16x3 accumulation.
// silu*mul epilogue -> LDS as GEMM2 A-fragments (hi/lo bf16).
// GEMM2 (mid@Wdown): each wave owns 256 output columns.
// ---------------------------------------------------------------------------
#define WMMA3(acc, ah, al, bh, bl)                                                        \
    acc = __builtin_amdgcn_wmma_f32_16x16x32_bf16(false, (ah).v, false, (bh).v, (short)0, \
                                                  acc, false, false);                     \
    acc = __builtin_amdgcn_wmma_f32_16x16x32_bf16(false, (ah).v, false, (bl).v, (short)0, \
                                                  acc, false, false);                     \
    acc = __builtin_amdgcn_wmma_f32_16x16x32_bf16(false, (al).v, false, (bh).v, (short)0, \
                                                  acc, false, false);

__launch_bounds__(256)
__global__ void moe_mlp_fused(const float* __restrict__ hidden,
                              const unsigned* __restrict__ w1f,   // gate_up frags
                              const unsigned* __restrict__ w2f,   // down frags
                              float* __restrict__ out) {
    // GEMM2 A-fragment store for the whole 32x768 intermediate (hi+lo): 98304 B
    __shared__ __align__(16) unsigned lds_mid[24 * 2 * 32 * 16];
    // GEMM1 A staging: 128-wide K slab as fragments (hi+lo): 16384 B
    __shared__ __align__(16) unsigned lds_a[4 * 2 * 32 * 16];

    const int tid     = threadIdx.x;
    const int w       = tid >> 5;     // wave id 0..7
    const int l       = tid & 31;     // lane
    const int laneSub = l & 15;
    const int laneHi  = l >> 4;       // 0/1
    const int t0      = blockIdx.x * 32;

    v8f accg[6][2], accu[6][2];
    #pragma unroll
    for (int i = 0; i < 6; ++i)
        #pragma unroll
        for (int ms = 0; ms < 2; ++ms)
            #pragma unroll
            for (int v = 0; v < 8; ++v) { accg[i][ms][v] = 0.f; accu[i][ms][v] = 0.f; }

    // ================= GEMM1: K loop over H in 128-wide rounds ==============
    #pragma unroll 1
    for (int r = 0; r < 16; ++r) {
        // ---- cooperative A staging: one fragment record per thread --------
        {
            int lane_r = tid & 31;
            int ms     = (tid >> 5) & 1;
            int ks     = tid >> 6;                       // 0..3
            int m      = ms * 16 + (lane_r & 15);
            int kb     = r * 128 + ks * 32 + ((lane_r >> 4) << 3);
            unsigned* dstA = lds_a + (size_t)(((ks * 2 + ms) * 32 + lane_r)) * 16;
            const float* srow = hidden + (size_t)(t0 + m) * HID;
            #pragma unroll
            for (int j = 0; j < 8; ++j) {
                int kk = (j < 4) ? (2 * j) : (16 + 2 * (j - 4));   // A layout
                float2 ab = *(const float2*)(srow + kb + kk);
                unsigned short h0, l0, h1, l1;
                split2(ab.x, h0, l0);
                split2(ab.y, h1, l1);
                dstA[j]     = (unsigned)h0 | ((unsigned)h1 << 16);
                dstA[j + 8] = (unsigned)l0 | ((unsigned)l1 << 16);
            }
        }
        __syncthreads();

        #pragma unroll
        for (int ks = 0; ks < 4; ++ks) {
            FragU aH[2], aL[2];
            #pragma unroll
            for (int ms = 0; ms < 2; ++ms) {
                const unsigned* p = lds_a + ((ks * 2 + ms) * 32 + l) * 16;
                aH[ms].q[0] = *(const int4*)(p);
                aH[ms].q[1] = *(const int4*)(p + 4);
                aL[ms].q[0] = *(const int4*)(p + 8);
                aL[ms].q[1] = *(const int4*)(p + 12);
            }
            const int kc = r * 4 + ks;                   // global k-chunk 0..63
            #pragma unroll
            for (int i = 0; i < 6; ++i) {
                const int ntg = w * 6 + i;               // gate n-tile (0..47)
                const int ntu = 48 + w * 6 + i;          // up n-tile  (48..95)
                FragU bgH, bgL, buH, buL;
                const unsigned* pg = w1f + ((size_t)(ntg * 64 + kc) * 32 + l) * 16;
                bgH.q[0] = *(const int4*)(pg);      bgH.q[1] = *(const int4*)(pg + 4);
                bgL.q[0] = *(const int4*)(pg + 8);  bgL.q[1] = *(const int4*)(pg + 12);
                const unsigned* pu = w1f + ((size_t)(ntu * 64 + kc) * 32 + l) * 16;
                buH.q[0] = *(const int4*)(pu);      buH.q[1] = *(const int4*)(pu + 4);
                buL.q[0] = *(const int4*)(pu + 8);  buL.q[1] = *(const int4*)(pu + 12);
                #pragma unroll
                for (int ms = 0; ms < 2; ++ms) {
                    WMMA3(accg[i][ms], aH[ms], aL[ms], bgH, bgL);
                    WMMA3(accu[i][ms], aH[ms], aL[ms], buH, buL);
                }
            }
        }
        __syncthreads();
    }

    // ============ silu(gate)*up -> LDS as GEMM2 A-fragments =================
    #pragma unroll
    for (int i = 0; i < 6; ++i) {
        const int nl   = i * 16 + laneSub;               // n within wave's 96 cols
        const int ks2  = w * 3 + (nl >> 5);              // GEMM2 k-chunk 0..23
        const int kk   = nl & 31;
        const int lh   = (kk >> 3) & 1;                  // A-layout lane half
        const int j2   = ((kk & 7) >> 1) + ((kk >> 4) << 2);
        const int half = kk & 1;
        #pragma unroll
        for (int ms = 0; ms < 2; ++ms) {
            #pragma unroll
            for (int v = 0; v < 8; ++v) {
                const int m = ms * 16 + v + 8 * laneHi;
                float g = accg[i][ms][v];
                float u = accu[i][ms][v];
                float x = g / (1.0f + __expf(-g)) * u;   // silu(g) * u
                unsigned short xh, xl;
                split2(x, xh, xl);
                const int lane2 = (m & 15) + (lh << 4);
                const int dw = ((ks2 * 2 + ms) * 32 + lane2) * 16 + j2;
                unsigned short* base = (unsigned short*)lds_mid;
                base[dw * 2 + half]       = xh;          // hi plane (dwords 0..7)
                base[(dw + 8) * 2 + half] = xl;          // lo plane (dwords 8..15)
            }
        }
    }
    __syncthreads();

    // ================= GEMM2: mid[32x768] @ Wdown[768x2048] ================
    #pragma unroll 1
    for (int p = 0; p < 8; ++p) {                        // 2 output n-tiles/pass
        v8f acc[2][2];
        #pragma unroll
        for (int pr = 0; pr < 2; ++pr)
            #pragma unroll
            for (int ms = 0; ms < 2; ++ms)
                #pragma unroll
                for (int v = 0; v < 8; ++v) acc[pr][ms][v] = 0.f;

        #pragma unroll 2
        for (int ks2 = 0; ks2 < 24; ++ks2) {
            FragU aH[2], aL[2];
            #pragma unroll
            for (int ms = 0; ms < 2; ++ms) {
                const unsigned* pa = lds_mid + ((ks2 * 2 + ms) * 32 + l) * 16;
                aH[ms].q[0] = *(const int4*)(pa);
                aH[ms].q[1] = *(const int4*)(pa + 4);
                aL[ms].q[0] = *(const int4*)(pa + 8);
                aL[ms].q[1] = *(const int4*)(pa + 12);
            }
            #pragma unroll
            for (int pr = 0; pr < 2; ++pr) {
                const int nt2 = w * 16 + p * 2 + pr;     // 0..127
                const unsigned* pb = w2f + ((size_t)(nt2 * 24 + ks2) * 32 + l) * 16;
                FragU bH, bL;
                bH.q[0] = *(const int4*)(pb);      bH.q[1] = *(const int4*)(pb + 4);
                bL.q[0] = *(const int4*)(pb + 8);  bL.q[1] = *(const int4*)(pb + 12);
                #pragma unroll
                for (int ms = 0; ms < 2; ++ms) {
                    WMMA3(acc[pr][ms], aH[ms], aL[ms], bH, bL);
                }
            }
        }
        // ---- fp32 output store (C layout: lane<16 M=v, lane>=16 M=v+8) ----
        #pragma unroll
        for (int pr = 0; pr < 2; ++pr) {
            const int col = (w * 16 + p * 2 + pr) * 16 + laneSub;
            #pragma unroll
            for (int ms = 0; ms < 2; ++ms)
                #pragma unroll
                for (int v = 0; v < 8; ++v) {
                    const int m = ms * 16 + v + 8 * laneHi;
                    out[(size_t)(t0 + m) * HID + col] = acc[pr][ms][v];
                }
        }
    }
}

// ---------------------------------------------------------------------------
extern "C" void kernel_launch(void* const* d_in, const int* in_sizes, int n_in,
                              void* d_out, int out_size, void* d_ws, size_t ws_size,
                              hipStream_t stream) {
    (void)in_sizes; (void)n_in; (void)out_size; (void)ws_size;
    const float* hidden    = (const float*)d_in[0];   // [32768, 2048]
    const float* w_gate_up = (const float*)d_in[1];   // [2048, 1536]
    const float* w_down    = (const float*)d_in[2];   // [768, 2048]
    float* out = (float*)d_out;

    unsigned* ws1 = (unsigned*)d_ws;                       // gate_up frags: 12.6 MB
    const size_t w1dw = (size_t)96 * 64 * 32 * 16;
    unsigned* ws2 = ws1 + w1dw;                            // down frags: 6.3 MB

    const int rec1 = 96 * 64 * 32;    // (1536/16) n-tiles x (2048/32) k-chunks x 32 lanes
    const int rec2 = 128 * 24 * 32;   // (2048/16) x (768/32) x 32

    convert_b_frags<<<(rec1 + 255) / 256, 256, 0, stream>>>(w_gate_up, ws1, 1536, 64, rec1);
    convert_b_frags<<<(rec2 + 255) / 256, 256, 0, stream>>>(w_down,    ws2, 2048, 24, rec2);
    moe_mlp_fused<<<T_TOK / 32, 256, 0, stream>>>(hidden, ws1, ws2, out);
}
